// FastSelfAttention_18820546691353
// MI455X (gfx1250) — matrix operand order repaired and checked by reading, not compiled, not run on hardware
//
#include <hip/hip_runtime.h>
#include <math.h>

// Problem constants (from reference)
#define NROWS 8192
#define DDIM  1024
#define NHEAD 16
#define HDIM  64
#define SPIN0 4096

typedef __bf16 bf16_t;
typedef bf16_t v16bf __attribute__((ext_vector_type(16)));
typedef bf16_t v8bf  __attribute__((ext_vector_type(8)));
typedef float  v8f   __attribute__((ext_vector_type(8)));
typedef int    v4i   __attribute__((ext_vector_type(4)));

// --- CDNA5 async global->LDS DMA (guarded; falls back to ld/st) -------------
#if defined(__HIP_DEVICE_COMPILE__) &&                                         \
    __has_builtin(__builtin_amdgcn_global_load_async_to_lds_b128) &&           \
    __has_builtin(__builtin_amdgcn_s_wait_asynccnt)
#define USE_ASYNC 1
#else
#define USE_ASYNC 0
#endif

#if USE_ASYNC
__device__ __forceinline__ void async_cp16(const bf16_t* g, bf16_t* l) {
  __builtin_amdgcn_global_load_async_to_lds_b128(
      (__attribute__((address_space(1))) v4i*)(g),
      (__attribute__((address_space(3))) v4i*)(l),
      /*offset=*/0, /*cpol=*/0);
}
#endif

// ---------------------------------------------------------------------------
// f32 -> bf16 conversion (grid-stride, row-major preserved)
// ---------------------------------------------------------------------------
__global__ void cvt_f32_bf16(const float* __restrict__ src,
                             bf16_t* __restrict__ dst, int n) {
  int i = blockIdx.x * blockDim.x + threadIdx.x;
  int stride = gridDim.x * blockDim.x;
  for (; i < n; i += stride) dst[i] = (bf16_t)src[i];
}

// ---------------------------------------------------------------------------
// f32 [K][N] -> bf16 transposed [N][K] (tiled LDS transpose, one-time cost)
// Grid (DDIM/32, DDIM/32), 256 threads.
// ---------------------------------------------------------------------------
__global__ __launch_bounds__(256)
void cvt_transpose_bf16(const float* __restrict__ w, bf16_t* __restrict__ wT) {
  __shared__ float tile[32][33];
  const int nb = blockIdx.x * 32;        // N base
  const int kb = blockIdx.y * 32;        // K base
  const int tx = threadIdx.x & 31;
  const int ty = (threadIdx.x >> 5) * 4; // 8 waves * 4 rows
#pragma unroll
  for (int j = 0; j < 4; ++j)
    tile[ty + j][tx] = w[(size_t)(kb + ty + j) * DDIM + nb + tx];
  __syncthreads();
#pragma unroll
  for (int j = 0; j < 4; ++j)
    wT[(size_t)(nb + ty + j) * DDIM + kb + tx] = (bf16_t)tile[tx][ty + j];
}

// ---------------------------------------------------------------------------
// bf16 WMMA GEMM: out[8192,1024] = A[M,K] @ W[K,N] + bias, W given TRANSPOSED
// as WT[N][K] so every LDS stage is a 16B vector copy (no scatter).
// BM=128 x BN=64 block tile, 8 waves, each wave owns a 32x32 output
// (2 A-frags x 2 B-frags -> 4 WMMAs per K-step).
// Double-buffered LDS staging via GLOBAL_LOAD_ASYNC_TO_LDS_B128: tile i+1
// streams under the WMMAs of tile i; async loads complete in order, so
// s_wait_asynccnt(3) (3 DMAs/tile/wave in flight) proves tile i has landed.
// Rows < SPIN0 use (WT0,b0); rows >= SPIN0 use (WT1,b1); BM | SPIN0.
// Epilogue: addsrc ? (acc+bias+addsrc)*outScale : acc+bias
// ---------------------------------------------------------------------------
#define BM 128
#define BN 64
#define BK 32
#define LP (BK + 8)   // row pitch 40 elems = 80B = 5*16B

__global__ __launch_bounds__(256)
void wmma_gemm(const bf16_t* __restrict__ A,
               const bf16_t* __restrict__ WT0, const bf16_t* __restrict__ WT1,
               const float* __restrict__ b0,  const float* __restrict__ b1,
               const float* __restrict__ addsrc, float outScale,
               float* __restrict__ out) {
  __shared__ __align__(16) bf16_t As[2][BM][LP];
  __shared__ __align__(16) bf16_t Bs[2][BN][LP];

  const int t    = threadIdx.x;
  const int lane = t & 31;
  const int wave = t >> 5;            // 8 waves
  const int wm   = (wave & 3) * 32;   // wave M offset in block tile
  const int wn   = (wave >> 2) * 32;  // wave N offset in block tile
  const int lh   = lane >> 4;         // K-half selector (ISA 16-bit layout)
  const int lm   = lane & 15;

  const int rowBase = blockIdx.y * BM;
  const int colBase = blockIdx.x * BN;

  const bf16_t* WT  = (rowBase < SPIN0) ? WT0 : WT1;
  const float* bias = (rowBase < SPIN0) ? b0 : b1;

  v8f acc[2][2] = {};

  // staging: A 128x32 -> 16 elems/thread (2x 16B); B 64x32 -> 8 elems/thread
  const int ar = t >> 1;
  const int ac = (t & 1) * 16;
  const int br = t >> 2;
  const int bc = (t & 3) * 8;

  const bf16_t* gA = A  + (size_t)(rowBase + ar) * DDIM + ac;
  const bf16_t* gB = WT + (size_t)(colBase + br) * DDIM + bc;

#if USE_ASYNC
  // prologue: stage tile 0 into buffer 0 (overlaps wave setup)
  async_cp16(gA,     &As[0][ar][ac]);
  async_cp16(gA + 8, &As[0][ar][ac + 8]);
  async_cp16(gB,     &Bs[0][br][bc]);
#endif

  for (int k0 = 0, buf = 0; k0 < DDIM; k0 += BK, buf ^= 1) {
#if USE_ASYNC
    if (k0 + BK < DDIM) {
      const int nb = buf ^ 1;
      async_cp16(gA + k0 + BK,     &As[nb][ar][ac]);
      async_cp16(gA + k0 + BK + 8, &As[nb][ar][ac + 8]);
      async_cp16(gB + k0 + BK,     &Bs[nb][br][bc]);
      __builtin_prefetch(gA + k0 + 2 * BK, 0, 3);   // warm L2 for tile i+2
      __builtin_prefetch(gB + k0 + 2 * BK, 0, 3);
      __builtin_amdgcn_s_wait_asynccnt(3);  // tile i landed; i+1 in flight
    } else {
      __builtin_amdgcn_s_wait_asynccnt(0);  // last tile: drain
    }
#else
    *(v8bf*)(&As[buf][ar][ac])     = *(const v8bf*)(gA + k0);
    *(v8bf*)(&As[buf][ar][ac + 8]) = *(const v8bf*)(gA + k0 + 8);
    *(v8bf*)(&Bs[buf][br][bc])     = *(const v8bf*)(gB + k0);
#endif
    __syncthreads();  // RAW: all waves' tile-i data visible

    // fragments per CDNA5 16-bit WMMA layouts
    union { v16bf v; v8bf h[2]; } a[2], b[2];
#pragma unroll
    for (int mi = 0; mi < 2; ++mi) {
      const int arow = wm + mi * 16 + lm;
      const int ab   = lh * 8;
      a[mi].h[0] = *(const v8bf*)(&As[buf][arow][ab]);
      a[mi].h[1] = *(const v8bf*)(&As[buf][arow][ab + 16]);
    }
#pragma unroll
    for (int ni = 0; ni < 2; ++ni) {
      const int bcol = wn + ni * 16 + lm;
      const int kb   = lh * 8;
      b[ni].h[0] = *(const v8bf*)(&Bs[buf][bcol][kb]);
      b[ni].h[1] = *(const v8bf*)(&Bs[buf][bcol][kb + 16]);
    }
#pragma unroll
    for (int mi = 0; mi < 2; ++mi)
#pragma unroll
      for (int ni = 0; ni < 2; ++ni)
        acc[mi][ni] = __builtin_amdgcn_wmma_f32_16x16x32_bf16(
            false, a[mi].v, false, b[ni].v, (short)0, acc[mi][ni],
            false, false);

    __syncthreads();  // WAR: buffer may be overwritten by next issue
  }

  // C/D layout: VGPR r -> M = r (lanes 0-15) / 8+r (lanes 16-31), N = lane&15
#pragma unroll
  for (int mi = 0; mi < 2; ++mi) {
    const int cmBase = rowBase + wm + mi * 16 + lh * 8;
#pragma unroll
    for (int ni = 0; ni < 2; ++ni) {
      const int cn = colBase + wn + ni * 16 + lm;
      const float bb = bias[cn];
#pragma unroll
      for (int r = 0; r < 8; ++r) {
        const int row = cmBase + r;
        float vo = acc[mi][ni][r] + bb;
        if (addsrc) vo = (vo + addsrc[(size_t)row * DDIM + cn]) * outScale;
        out[(size_t)row * DDIM + cn] = vo;
      }
    }
  }
}

// ---------------------------------------------------------------------------
// score = (rows * scale?) @ w[D,H] + b[H]   one thread per (n,h)
// ---------------------------------------------------------------------------
__global__ void score_kernel(const float* __restrict__ rows,
                             const float* __restrict__ scale,
                             const float* __restrict__ w,
                             const float* __restrict__ b,
                             float* __restrict__ score) {
  int idx = blockIdx.x * blockDim.x + threadIdx.x;
  if (idx >= NROWS * NHEAD) return;
  int n = idx / NHEAD, h = idx % NHEAD;
  const float* r = rows + (size_t)n * DDIM;
  float s = b[h];
  if (scale) {
    for (int d = 0; d < DDIM; ++d) s += r[d] * scale[d] * w[d * NHEAD + h];
  } else {
    for (int d = 0; d < DDIM; ++d) s += r[d] * w[d * NHEAD + h];
  }
  score[idx] = s;
}

// ---------------------------------------------------------------------------
// per-head softmax stats over N: stats[h]=max, stats[H+h]=sum(exp(x-max))
// ---------------------------------------------------------------------------
__global__ __launch_bounds__(256)
void softmax_stats(const float* __restrict__ score, float* __restrict__ stats) {
  const int h = blockIdx.x;
  const int t = threadIdx.x;
  __shared__ float red[256];
  float m = -3.4e38f;
  for (int n = t; n < NROWS; n += 256) m = fmaxf(m, score[n * NHEAD + h]);
  red[t] = m; __syncthreads();
  for (int s = 128; s > 0; s >>= 1) {
    if (t < s) red[t] = fmaxf(red[t], red[t + s]);
    __syncthreads();
  }
  m = red[0]; __syncthreads();
  float sum = 0.f;
  for (int n = t; n < NROWS; n += 256) sum += __expf(score[n * NHEAD + h] - m);
  red[t] = sum; __syncthreads();
  for (int s = 128; s > 0; s >>= 1) {
    if (t < s) red[t] += red[t + s];
    __syncthreads();
  }
  if (t == 0) { stats[h] = m; stats[NHEAD + h] = red[0]; }
}

__global__ void zero_f32(float* p, int n) {
  int i = blockIdx.x * blockDim.x + threadIdx.x;
  if (i < n) p[i] = 0.f;
}

// ---------------------------------------------------------------------------
// pooled[h*HD+hd] += sum_n softmax(score)[h,n] * rows[n, h*HD+hd]
// grid = (NHEAD, 8 N-chunks); block 256 = 4 n-groups x 64 hd
// ---------------------------------------------------------------------------
__global__ __launch_bounds__(256)
void pooled_accum(const float* __restrict__ score, const float* __restrict__ stats,
                  const float* __restrict__ rows, float* __restrict__ pooled) {
  const int h  = blockIdx.x;
  const int n0 = blockIdx.y * (NROWS / 8);
  const int t  = threadIdx.x;
  const int hd = t & 63;
  const int g  = t >> 6;
  const float m   = stats[h];
  const float inv = 1.0f / stats[NHEAD + h];
  float acc = 0.f;
  for (int n = n0 + g; n < n0 + NROWS / 8; n += 4) {
    float wgt = __expf(score[n * NHEAD + h] - m) * inv;
    acc += wgt * rows[(size_t)n * DDIM + h * HDIM + hd];
  }
  __shared__ float part[256];
  part[t] = acc; __syncthreads();
  if (g == 0)
    atomicAdd(&pooled[h * HDIM + hd],
              part[hd] + part[64 + hd] + part[128 + hd] + part[192 + hd]);
}

// ---------------------------------------------------------------------------
// dst_bf16[n,d] = bf16(v[n,d] * pooled[d])   (pre-scale A for final GEMM)
// ---------------------------------------------------------------------------
__global__ void scale_cvt(const float* __restrict__ v,
                          const float* __restrict__ pooled,
                          bf16_t* __restrict__ dst) {
  int i = blockIdx.x * blockDim.x + threadIdx.x;
  int stride = gridDim.x * blockDim.x;
  for (; i < NROWS * DDIM; i += stride) {
    int d = i & (DDIM - 1);
    dst[i] = (bf16_t)(v[i] * pooled[d]);
  }
}

// ---------------------------------------------------------------------------
extern "C" void kernel_launch(void* const* d_in, const int* in_sizes, int n_in,
                              void* d_out, int out_size, void* d_ws, size_t ws_size,
                              hipStream_t stream) {
  (void)in_sizes; (void)n_in; (void)out_size; (void)ws_size;

  const float* hs  = (const float*)d_in[0];
  const float* q0w = (const float*)d_in[1];
  const float* q0b = (const float*)d_in[2];
  const float* q1w = (const float*)d_in[3];
  const float* q1b = (const float*)d_in[4];
  const float* k0w = (const float*)d_in[5];
  const float* k0b = (const float*)d_in[6];
  const float* k1w = (const float*)d_in[7];
  const float* k1b = (const float*)d_in[8];
  const float* v0w = (const float*)d_in[9];
  const float* v0b = (const float*)d_in[10];
  const float* v1w = (const float*)d_in[11];
  const float* v1b = (const float*)d_in[12];
  const float* wqw = (const float*)d_in[13];
  const float* wqb = (const float*)d_in[14];
  const float* wkw = (const float*)d_in[15];
  const float* wkb = (const float*)d_in[16];
  const float* tw  = (const float*)d_in[17];
  const float* tb  = (const float*)d_in[18];

  size_t off = 0;
  char* base = (char*)d_ws;
  auto carve = [&](size_t bytes) -> void* {
    void* p = base + off;
    off += (bytes + 255) & ~(size_t)255;
    return p;
  };
  bf16_t* hsb  = (bf16_t*)carve((size_t)NROWS * DDIM * 2);
  bf16_t* wq0  = (bf16_t*)carve((size_t)DDIM * DDIM * 2);  // all weights stored transposed [N][K]
  bf16_t* wq1  = (bf16_t*)carve((size_t)DDIM * DDIM * 2);
  bf16_t* wk0  = (bf16_t*)carve((size_t)DDIM * DDIM * 2);
  bf16_t* wk1  = (bf16_t*)carve((size_t)DDIM * DDIM * 2);
  bf16_t* wv0  = (bf16_t*)carve((size_t)DDIM * DDIM * 2);
  bf16_t* wv1  = (bf16_t*)carve((size_t)DDIM * DDIM * 2);
  bf16_t* wtT  = (bf16_t*)carve((size_t)DDIM * DDIM * 2);
  float*  qf   = (float*)carve((size_t)NROWS * DDIM * 4);
  float*  kf   = (float*)carve((size_t)NROWS * DDIM * 4);
  float*  vf   = (float*)carve((size_t)NROWS * DDIM * 4);
  bf16_t* wvb  = (bf16_t*)carve((size_t)NROWS * DDIM * 2);
  float*  score   = (float*)carve((size_t)NROWS * NHEAD * 4);
  float*  stats   = (float*)carve((size_t)2 * NHEAD * 4);
  float*  pooledq = (float*)carve((size_t)DDIM * 4);
  float*  pooledk = (float*)carve((size_t)DDIM * 4);

  // 1) convert activations (row-major) + weights (transposed) to bf16
  dim3 tg(DDIM / 32, DDIM / 32);
  cvt_f32_bf16<<<2048, 256, 0, stream>>>(hs, hsb, NROWS * DDIM);
  cvt_transpose_bf16<<<tg, 256, 0, stream>>>(q0w, wq0);
  cvt_transpose_bf16<<<tg, 256, 0, stream>>>(q1w, wq1);
  cvt_transpose_bf16<<<tg, 256, 0, stream>>>(k0w, wk0);
  cvt_transpose_bf16<<<tg, 256, 0, stream>>>(k1w, wk1);
  cvt_transpose_bf16<<<tg, 256, 0, stream>>>(v0w, wv0);
  cvt_transpose_bf16<<<tg, 256, 0, stream>>>(v1w, wv1);
  cvt_transpose_bf16<<<tg, 256, 0, stream>>>(tw,  wtT);

  // 2) q/k/v GEMMs
  dim3 gg(DDIM / BN, NROWS / BM);
  wmma_gemm<<<gg, 256, 0, stream>>>(hsb, wq0, wq1, q0b, q1b, nullptr, 1.0f, qf);
  wmma_gemm<<<gg, 256, 0, stream>>>(hsb, wk0, wk1, k0b, k1b, nullptr, 1.0f, kf);
  wmma_gemm<<<gg, 256, 0, stream>>>(hsb, wv0, wv1, v0b, v1b, nullptr, 1.0f, vf);

  // 3) q attention pooling
  score_kernel<<<(NROWS * NHEAD) / 256, 256, 0, stream>>>(qf, nullptr, wqw, wqb, score);
  softmax_stats<<<NHEAD, 256, 0, stream>>>(score, stats);
  zero_f32<<<DDIM / 256, 256, 0, stream>>>(pooledq, DDIM);
  pooled_accum<<<dim3(NHEAD, 8), 256, 0, stream>>>(score, stats, qf, pooledq);

  // 4) k attention pooling (k pre-scaled by pooled_q inside score)
  score_kernel<<<(NROWS * NHEAD) / 256, 256, 0, stream>>>(kf, pooledq, wkw, wkb, score);
  softmax_stats<<<NHEAD, 256, 0, stream>>>(score, stats);
  zero_f32<<<DDIM / 256, 256, 0, stream>>>(pooledk, DDIM);
  pooled_accum<<<dim3(NHEAD, 8), 256, 0, stream>>>(score, stats, kf, pooledk);

  // 5) weighted_v in bf16, then final GEMM with (+t_b +q)/sqrt(2) epilogue
  scale_cvt<<<2048, 256, 0, stream>>>(vf, pooledk, wvb);
  wmma_gemm<<<gg, 256, 0, stream>>>(wvb, wtT, wtT, tb, tb, qf,
                                    0.70710678118654752f, (float*)d_out);
}
